// EvoformerBlock_13443247636738
// MI455X (gfx1250) — matrix operand order repaired and checked by reading, not compiled, not run on hardware
//
#include <hip/hip_runtime.h>

typedef __bf16 bf16_t;
typedef __attribute__((ext_vector_type(16))) __bf16 v16b;
typedef __attribute__((ext_vector_type(8)))  float  v8f;

// ---------------------------------------------------------------------------
// Generic WMMA GEMM:  C = epilogue(alpha * op(A) @ op(B))
//   AM: 0 = A bf16 [M,K] (lda=row stride)
//       1 = A bf16 [K,M] (transposed)
//       2 = A bf16 OPM remap: row=(i*256+j), k=(c*32+d) -> A[(i*32+c)*8192 + j*32+d]
//   BM: 0 = B f32  [K,N]   (weights, converted to bf16 while staging)
//       1 = B bf16 [K,N]
//       2 = B bf16 [N,K]   (transposed -> A@B^T)
//   ACT: 0 none, 1 relu, 2 sigmoid
// Epilogue order: v = alpha*acc; +bias[col]; act; *emul[r,c]; *rowscale[r]; +resid[r,c]
// Outputs: optional f32 (Cf, strides cfr/cfc) and/or bf16 (Cb, strides cbr/cbc).
// Batched over blockIdx.z with element strides sA/sB/sC.
// B tile is kept TRANSPOSED in LDS (BsT[col][k]) so every lane's B-fragment is
// two contiguous ds_load_b128 reads (no per-element gather before the WMMA).
// ---------------------------------------------------------------------------
struct GemmP {
  const bf16_t* A;
  const float*  Bf;
  const bf16_t* Bb;
  const float*  bias;
  const float*  emul;
  const float*  rowscale;
  const float*  resid;
  float*   Cf;
  bf16_t*  Cb;
  int M, N, K, lda, ldb;
  float alpha;
  long cfr, cfc, cbr, cbc;
  long sA, sB, sC;
};

template<int AM, int BM, int ACT>
__global__ __launch_bounds__(128) void gemm_k(GemmP p) {
  __shared__ bf16_t As [64][40];   // [row][k]  row stride 80B (16B aligned)
  __shared__ bf16_t BsT[64][40];   // [col][k]  row stride 80B (16B aligned)

  const int z = blockIdx.z;
  const bf16_t* Ap  = p.A + (size_t)z * p.sA;
  const float*  Bfp = (BM == 0) ? (p.Bf + (size_t)z * p.sB) : nullptr;
  const bf16_t* Bbp = (BM != 0) ? (p.Bb + (size_t)z * p.sB) : nullptr;

  const int row0 = blockIdx.y * 64;
  const int col0 = blockIdx.x * 64;
  const int t    = threadIdx.x;
  const int wv   = t >> 5, lane = t & 31;
  const int wy   = (wv >> 1) * 32, wx = (wv & 1) * 32;
  const int l15  = lane & 15, lh = lane >> 4;

  v8f acc[2][2] = {};

  for (int kb = 0; kb < p.K; kb += 32) {
    __syncthreads();
    // ---------------- stage A tile (64 rows x 32 k) ----------------
    if (AM == 1) {
      const int k = t >> 2, seg = (t & 3) * 16;
      bf16_t tmp[16];
      if (row0 + seg < p.M) {
        const bf16_t* src = Ap + (size_t)(kb + k) * p.lda + row0 + seg;
        *(uint4*)&tmp[0] = *(const uint4*)src;
        *(uint4*)&tmp[8] = *(const uint4*)(src + 8);
      } else {
#pragma unroll
        for (int e = 0; e < 16; ++e) tmp[e] = (bf16_t)0.f;
      }
#pragma unroll
      for (int e = 0; e < 16; ++e) As[seg + e][k] = tmp[e];
    } else {
      const int r = t >> 1, seg = (t & 1) * 16;
      const int gr = row0 + r;
      uint4 v0 = {0, 0, 0, 0}, v1 = {0, 0, 0, 0};
      if (gr < p.M) {
        const bf16_t* src;
        if (AM == 0) {
          src = Ap + (size_t)gr * p.lda + kb + seg;
        } else {  // AM == 2, kb is a multiple of 32 -> contiguous d-run
          const int i = gr >> 8, j = gr & 255, c = kb >> 5;
          src = Ap + (((size_t)(i * 32 + c)) << 13) + (size_t)(j * 32) + seg;
        }
        v0 = *(const uint4*)src;
        v1 = *(const uint4*)(src + 8);
        if (kb + 32 < p.K)
          __builtin_prefetch(AM == 0 ? (const void*)(src + 32) : (const void*)(src + 8192));
      }
      *(uint4*)&As[r][seg]     = v0;
      *(uint4*)&As[r][seg + 8] = v1;
    }
    // ---------------- stage B tile (32 k x 64 cols) into BsT[col][k] ----------------
    if (BM == 2) {
      // global is k-contiguous: vectorized on both sides
      const int n = t >> 1, seg = (t & 1) * 16;
      const int gn = col0 + n;
      uint4 v0 = {0, 0, 0, 0}, v1 = {0, 0, 0, 0};
      if (gn < p.N) {
        const bf16_t* src = Bbp + (size_t)gn * p.ldb + kb + seg;
        v0 = *(const uint4*)src;
        v1 = *(const uint4*)(src + 8);
        if (kb + 32 < p.K) __builtin_prefetch((const void*)(src + 32));
      }
      *(uint4*)&BsT[n][seg]     = v0;
      *(uint4*)&BsT[n][seg + 8] = v1;
    } else {
      // global is n-contiguous: vector load, transpose-scatter into LDS
      const int k = t >> 2, seg = (t & 3) * 16;
      const int gn0 = col0 + seg;
      bf16_t tmp[16];
      if (gn0 < p.N) {
        if (BM == 0) {
          const float* src = Bfp + (size_t)(kb + k) * p.ldb + gn0;
          const float4* s4 = (const float4*)src;
#pragma unroll
          for (int q = 0; q < 4; ++q) {
            const float4 f = s4[q];
            tmp[q * 4 + 0] = (bf16_t)f.x; tmp[q * 4 + 1] = (bf16_t)f.y;
            tmp[q * 4 + 2] = (bf16_t)f.z; tmp[q * 4 + 3] = (bf16_t)f.w;
          }
          if (kb + 32 < p.K) __builtin_prefetch((const void*)(src + (size_t)32 * p.ldb));
        } else {  // BM == 1
          const bf16_t* src = Bbp + (size_t)(kb + k) * p.ldb + gn0;
          *(uint4*)&tmp[0] = *(const uint4*)src;
          *(uint4*)&tmp[8] = *(const uint4*)(src + 8);
          if (kb + 32 < p.K) __builtin_prefetch((const void*)(src + (size_t)32 * p.ldb));
        }
      } else {
#pragma unroll
        for (int e = 0; e < 16; ++e) tmp[e] = (bf16_t)0.f;
      }
#pragma unroll
      for (int e = 0; e < 16; ++e) BsT[seg + e][k] = tmp[e];
    }
    __syncthreads();

    v16b af[2], bfg[2];
#pragma unroll
    for (int ti = 0; ti < 2; ++ti) {
      const bf16_t* ar = &As[wy + ti * 16 + l15][0];
      union { v16b v; uint4 q[2]; } u;
      u.q[0] = *(const uint4*)(ar + lh * 8);        // K = lh*8 .. +7
      u.q[1] = *(const uint4*)(ar + 16 + lh * 8);   // K = 16+lh*8 .. +7
      af[ti] = u.v;
    }
#pragma unroll
    for (int tj = 0; tj < 2; ++tj) {
      const bf16_t* br = &BsT[wx + tj * 16 + l15][lh * 8];  // K = lh*8 .. +7, 16+lh*8 .. +7
      union { v16b v; uint4 q[2]; } u;
      u.q[0] = *(const uint4*)br;
      u.q[1] = *(const uint4*)(br + 16);
      bfg[tj] = u.v;
    }
#pragma unroll
    for (int ti = 0; ti < 2; ++ti)
#pragma unroll
      for (int tj = 0; tj < 2; ++tj)
        acc[ti][tj] = __builtin_amdgcn_wmma_f32_16x16x32_bf16(
            false, af[ti], false, bfg[tj], (short)0, acc[ti][tj], false, false);
  }

  float*  Cfp = p.Cf ? p.Cf + (size_t)z * p.sC : nullptr;
  bf16_t* Cbp = p.Cb ? p.Cb + (size_t)z * p.sC : nullptr;
#pragma unroll
  for (int ti = 0; ti < 2; ++ti)
#pragma unroll
    for (int tj = 0; tj < 2; ++tj)
#pragma unroll
      for (int i = 0; i < 8; ++i) {
        const int r = row0 + wy + ti * 16 + lh * 8 + i;
        const int c = col0 + wx + tj * 16 + l15;
        if (r < p.M && c < p.N) {
          float v = acc[ti][tj][i] * p.alpha;
          if (p.bias) v += p.bias[c];
          if (ACT == 1) v = fmaxf(v, 0.f);
          if (ACT == 2) v = 1.f / (1.f + __expf(-v));
          if (p.emul)     v *= p.emul[(size_t)r * p.N + c];
          if (p.rowscale) v *= p.rowscale[r];
          if (p.resid)    v += p.resid[(size_t)r * p.N + c];
          if (Cfp) Cfp[(size_t)r * p.cfr + (size_t)c * p.cfc] = v;
          if (Cbp) Cbp[(size_t)r * p.cbr + (size_t)c * p.cbc] = (bf16_t)v;
        }
      }
}

// ---------------------------------------------------------------------------
// Fused gated multi-head attention (one wave per 16 query rows of one (b,h)).
// Q pre-scaled by 1/sqrt(32). Head dim fixed at 32. Bias/mask/softmax handled
// in the LDS phase (contiguous per-row key walks). V staged transposed in LDS
// so PV fragments are vector LDS reads.
// ---------------------------------------------------------------------------
template<int HASPB>
__global__ __launch_bounds__(32) void attn_k(const bf16_t* __restrict__ Q,
                                             const bf16_t* __restrict__ Kx,
                                             const bf16_t* __restrict__ V,
                                             const float*  __restrict__ G,
                                             bf16_t* __restrict__ O,
                                             const float* __restrict__ mask,
                                             long msb, long msk,
                                             const float* __restrict__ pbias,
                                             int H, int L) {
  __shared__ float  slog[16][256];
  __shared__ bf16_t sp[16][256];
  __shared__ bf16_t VsT[32][256];

  const int b = blockIdx.z, h = blockIdx.y, q0 = blockIdx.x * 16;
  const int HC = H * 32;
  const size_t base = (size_t)b * L * HC + (size_t)h * 32;
  const bf16_t* Qp = Q + base;
  const bf16_t* Kp = Kx + base;
  const bf16_t* Vp = V + base;
  const int lane = threadIdx.x, l15 = lane & 15, lh = lane >> 4;

  // stage V transposed: VsT[c][r] = V[r][c]
  for (int r = lane; r < L; r += 32) {
    union { uint4 q[4]; bf16_t e[32]; } u;
    const uint4* vr = (const uint4*)(Vp + (size_t)r * HC);
    u.q[0] = vr[0]; u.q[1] = vr[1]; u.q[2] = vr[2]; u.q[3] = vr[3];
#pragma unroll
    for (int c = 0; c < 32; ++c) VsT[c][r] = u.e[c];
  }

  v16b qf;
  {
    const bf16_t* qr = Qp + (size_t)(q0 + l15) * HC + lh * 8;
    union { v16b v; uint4 q[2]; } u;
    u.q[0] = *(const uint4*)qr;
    u.q[1] = *(const uint4*)(qr + 16);
    qf = u.v;
  }

  for (int kt = 0; kt < L / 16; ++kt) {
    const int key = kt * 16 + l15;
    union { v16b v; uint4 q[2]; } uk;
    const uint4* kr = (const uint4*)(Kp + (size_t)key * HC + lh * 8);
    uk.q[0] = kr[0]; uk.q[1] = kr[2];
    v8f s = {};
    s = __builtin_amdgcn_wmma_f32_16x16x32_bf16(false, qf, false, uk.v, (short)0, s, false, false);
#pragma unroll
    for (int i = 0; i < 8; ++i) slog[lh * 8 + i][key] = s[i];
  }
  __syncthreads();

  {  // bias + mask + softmax; lane -> row l15, half lh of the key range
    const int r = l15, k0 = lh * (L / 2), k1 = k0 + L / 2;
    const float* pbr = HASPB ? (pbias + ((size_t)h * L + (q0 + r)) * L) : nullptr;
    float mx = -1e30f;
    for (int k = k0; k < k1; ++k) {
      float v = slog[r][k] + 1e9f * (mask[(size_t)b * msb + (size_t)k * msk] - 1.f);
      if (HASPB) v += pbr[k];
      slog[r][k] = v;
      mx = fmaxf(mx, v);
    }
    mx = fmaxf(mx, __shfl_xor(mx, 16, 32));
    float sum = 0.f;
    for (int k = k0; k < k1; ++k) { const float e = __expf(slog[r][k] - mx); slog[r][k] = e; sum += e; }
    sum += __shfl_xor(sum, 16, 32);
    const float inv = 1.f / sum;
    for (int k = k0; k < k1; ++k) sp[r][k] = (bf16_t)(slog[r][k] * inv);
  }
  __syncthreads();

  for (int ct = 0; ct < 2; ++ct) {
    v8f o = {};
    const int cn = ct * 16 + l15;
    for (int kb = 0; kb < L; kb += 32) {
      union { v16b v; uint4 q[2]; } up, uv;
      const bf16_t* pr = &sp[l15][kb + lh * 8];
      up.q[0] = *(const uint4*)pr;
      up.q[1] = *(const uint4*)(pr + 16);
      const bf16_t* vr2 = &VsT[cn][kb + lh * 8];
      uv.q[0] = *(const uint4*)vr2;
      uv.q[1] = *(const uint4*)(vr2 + 16);
      o = __builtin_amdgcn_wmma_f32_16x16x32_bf16(false, up.v, false, uv.v, (short)0, o, false, false);
    }
#pragma unroll
    for (int i = 0; i < 8; ++i) {
      const int r = q0 + lh * 8 + i;
      const int c = h * 32 + cn;
      const size_t gi = ((size_t)b * L + r) * HC + c;
      O[gi] = (bf16_t)((float)o[i] * G[gi]);
    }
  }
}

// LayerNorm over last dim (C in {128,256}); one wave per row; bf16 out.
__global__ __launch_bounds__(256) void ln_k(const float* __restrict__ x,
                                            const float* __restrict__ w,
                                            const float* __restrict__ b,
                                            bf16_t* __restrict__ out, int M, int C) {
  const int wid = threadIdx.x >> 5, lane = threadIdx.x & 31;
  const int row = blockIdx.x * 8 + wid;
  if (row >= M) return;
  const float* xr = x + (size_t)row * C;
  float s = 0.f, s2 = 0.f;
  for (int c = lane; c < C; c += 32) { const float v = xr[c]; s += v; s2 += v * v; }
#pragma unroll
  for (int o = 16; o > 0; o >>= 1) { s += __shfl_xor(s, o, 32); s2 += __shfl_xor(s2, o, 32); }
  const float mu = s / C;
  const float inv = rsqrtf(s2 / C - mu * mu + 1e-5f);
  bf16_t* orow = out + (size_t)row * C;
  for (int c = lane; c < C; c += 32) orow[c] = (bf16_t)((xr[c] - mu) * inv * w[c] + b[c]);
}

// Tiny projection z[M,K](bf16) @ wb[K,H](f32) stored head-major: out[h*M + row].
__global__ __launch_bounds__(256) void bias_proj_k(const bf16_t* __restrict__ z,
                                                   const float* __restrict__ wb,
                                                   float* __restrict__ out, int M, int K, int H) {
  const int row = blockIdx.x * 256 + threadIdx.x;
  if (row >= M) return;
  float acc[8];
  for (int h = 0; h < 8; ++h) acc[h] = 0.f;
  const bf16_t* zr = z + (size_t)row * K;
  for (int k = 0; k < K; ++k) {
    const float zv = (float)zr[k];
    for (int h = 0; h < H; ++h) acc[h] += zv * wb[k * H + h];
  }
  for (int h = 0; h < H; ++h) out[(size_t)h * M + row] = acc[h];
}

// rs[i*N+j] = 1/(1e-3 + sum_s mask[s,i]*mask[s,j])
__global__ __launch_bounds__(256) void opm_norm_k(const float* __restrict__ mask,
                                                  float* __restrict__ rs, int S, int N) {
  const int row = blockIdx.x * 256 + threadIdx.x;
  if (row >= N * N) return;
  const int i = row / N, j = row % N;
  float s = 0.f;
  for (int t = 0; t < S; ++t) s += mask[t * N + i] * mask[t * N + j];
  rs[row] = 1.f / (1e-3f + s);
}

// dst[j,i,c] = src[i,j,c]   (src: [A,B,C])
__global__ __launch_bounds__(256) void perm_k(const float* __restrict__ src,
                                              float* __restrict__ dst, int A, int Bd, int C) {
  const size_t idx = (size_t)blockIdx.x * 256 + threadIdx.x;
  const size_t tot = (size_t)A * Bd * C;
  if (idx >= tot) return;
  const int c = (int)(idx % C);
  const size_t ab = idx / C;
  const int j = (int)(ab % Bd);
  const int i = (int)(ab / Bd);
  dst[((size_t)j * A + i) * C + c] = src[idx];
}

// ---------------------------------------------------------------------------
// Host-side helpers
// ---------------------------------------------------------------------------
static GemmP mk(const bf16_t* A, const float* Bf, const bf16_t* Bb,
                int M, int N, int K, int lda, int ldb) {
  GemmP p{};
  p.A = A; p.Bf = Bf; p.Bb = Bb;
  p.M = M; p.N = N; p.K = K; p.lda = lda; p.ldb = ldb;
  p.alpha = 1.f;
  p.cfr = N; p.cfc = 1; p.cbr = N; p.cbc = 1;
  p.sA = 0; p.sB = 0; p.sC = 0;
  return p;
}
static dim3 ggrid(const GemmP& p, int batch) {
  return dim3((p.N + 63) / 64, (p.M + 63) / 64, batch);
}
static void g_nn(const GemmP& p, int act, hipStream_t s) {
  if (act == 0)      gemm_k<0, 0, 0><<<ggrid(p, 1), 128, 0, s>>>(p);
  else if (act == 1) gemm_k<0, 0, 1><<<ggrid(p, 1), 128, 0, s>>>(p);
  else               gemm_k<0, 0, 2><<<ggrid(p, 1), 128, 0, s>>>(p);
}
static void g_atb(const GemmP& p, int batch, hipStream_t s) { gemm_k<1, 1, 0><<<ggrid(p, batch), 128, 0, s>>>(p); }
static void g_abt(const GemmP& p, int batch, hipStream_t s) { gemm_k<0, 2, 0><<<ggrid(p, batch), 128, 0, s>>>(p); }
static void g_opm(const GemmP& p, hipStream_t s)            { gemm_k<2, 0, 0><<<ggrid(p, 1), 128, 0, s>>>(p); }
static void run_ln(const float* x, const float* w, const float* b, bf16_t* out,
                   int M, int C, hipStream_t s) {
  ln_k<<<(M + 7) / 8, 256, 0, s>>>(x, w, b, out, M, C);
}

extern "C" void kernel_launch(void* const* d_in, const int* in_sizes, int n_in,
                              void* d_out, int out_size, void* d_ws, size_t ws_size,
                              hipStream_t stream) {
  (void)n_in; (void)out_size; (void)ws_size;
  constexpr int S = 128, N = 256, CM = 256, CZ = 128;
  constexpr size_t MSA_E = (size_t)S * N * CM;
  const float QS = 0.17677669529663687f;  // 1/sqrt(32)

  // ---- top-level inputs (order auto-detected: insertion vs sorted) ----
  const float *msa_in, *pair_in, *msa_mask, *pair_mask;
  if (in_sizes[1] == S * N) {  // sorted: msa, msa_mask, pair, pair_mask
    msa_in = (const float*)d_in[0]; msa_mask = (const float*)d_in[1];
    pair_in = (const float*)d_in[2]; pair_mask = (const float*)d_in[3];
  } else {                     // insertion: msa, pair, msa_mask, pair_mask
    msa_in = (const float*)d_in[0]; pair_in = (const float*)d_in[1];
    msa_mask = (const float*)d_in[2]; pair_mask = (const float*)d_in[3];
  }

  // ---- params: JAX pytree (sorted-key) flattening ----
  int pi = 4;
  auto nx = [&]() { return (const float*)d_in[pi++]; };
  const float *col_bg=nx(), *col_bo=nx(), *col_lnb=nx(), *col_lnw=nx(),
              *col_wg=nx(), *col_wk=nx(), *col_wo=nx(), *col_wq=nx(), *col_wv=nx();
  const float *mt_b1=nx(), *mt_b2=nx(), *mt_lnb=nx(), *mt_lnw=nx(), *mt_w1=nx(), *mt_w2=nx();
  const float *op_b1=nx(), *op_b2=nx(), *op_bo=nx(), *op_lnb=nx(), *op_lnw=nx(),
              *op_w1=nx(), *op_w2=nx(), *op_wo=nx();
  const float *pt_b1=nx(), *pt_b2=nx(), *pt_lnb=nx(), *pt_lnw=nx(), *pt_w1=nx(), *pt_w2=nx();
  const float *row_bg=nx(), *row_bo=nx(), *row_lnb=nx(), *row_lnw=nx(),
              *row_lnzb=nx(), *row_lnzw=nx(), *row_wb=nx(),
              *row_wg=nx(), *row_wk=nx(), *row_wo=nx(), *row_wq=nx(), *row_wv=nx();
  const float *tae_bg=nx(), *tae_bo=nx(), *tae_lnb=nx(), *tae_lnw=nx(), *tae_wb=nx(),
              *tae_wg=nx(), *tae_wk=nx(), *tae_wo=nx(), *tae_wq=nx(), *tae_wv=nx();
  const float *tas_bg=nx(), *tas_bo=nx(), *tas_lnb=nx(), *tas_lnw=nx(), *tas_wb=nx(),
              *tas_wg=nx(), *tas_wk=nx(), *tas_wo=nx(), *tas_wq=nx(), *tas_wv=nx();
  const float *tmi_bag=nx(), *tmi_bap=nx(), *tmi_bbg=nx(), *tmi_bbp=nx(), *tmi_bg=nx(), *tmi_bz=nx(),
              *tmi_lnb=nx(), *tmi_lnw=nx(), *tmi_lnob=nx(), *tmi_lnow=nx(),
              *tmi_wag=nx(), *tmi_wap=nx(), *tmi_wbg=nx(), *tmi_wbp=nx(), *tmi_wg=nx(), *tmi_wz=nx();
  const float *tmo_bag=nx(), *tmo_bap=nx(), *tmo_bbg=nx(), *tmo_bbp=nx(), *tmo_bg=nx(), *tmo_bz=nx(),
              *tmo_lnb=nx(), *tmo_lnw=nx(), *tmo_lnob=nx(), *tmo_lnow=nx(),
              *tmo_wag=nx(), *tmo_wap=nx(), *tmo_wbg=nx(), *tmo_wbp=nx(), *tmo_wg=nx(), *tmo_wz=nx();

  // ---- workspace layout ----
  char* w = (char*)d_ws;
  bf16_t* W0  = (bf16_t*)(w);                         // 16 MiB : LN outputs
  bf16_t* W1  = (bf16_t*)(w + ((size_t)16 << 20));    // 16 MiB : Q / OPM a
  bf16_t* W2  = (bf16_t*)(w + ((size_t)32 << 20));    // 16 MiB : K / OPM b
  bf16_t* W3  = (bf16_t*)(w + ((size_t)48 << 20));    // 16 MiB : V
  float*  W4  = (float*) (w + ((size_t)64 << 20));    // 32 MiB : sigmoid gate
  bf16_t* W5  = (bf16_t*)(w + ((size_t)96 << 20));    // 16 MiB : O / zln / xln
  float*  W6  = (float*) (w + ((size_t)112 << 20));   // 32 MiB : transposed buf / tmp
  float*  W7  = (float*) (w + ((size_t)144 << 20));   // 32 MiB : tmp2
  bf16_t* W8  = (bf16_t*)(w + ((size_t)176 << 20));   // 128 MiB: outer2 / hidden / planes
  bf16_t* W8b = (bf16_t*)(w + ((size_t)192 << 20));   // second plane set (inside W8)
  float*  W9  = (float*) (w + ((size_t)304 << 20));   // 2 MiB  : pair bias [H,L,L]
  float*  Wrs = (float*) (w + ((size_t)306 << 20));   // 256 KiB: OPM norm

  float* msa  = (float*)d_out;           // [S*N, CM]
  float* pair = (float*)d_out + MSA_E;   // [N*N, CZ]
  hipMemcpyAsync(msa,  msa_in,  MSA_E * 4, hipMemcpyDeviceToDevice, stream);
  hipMemcpyAsync(pair, pair_in, MSA_E * 4, hipMemcpyDeviceToDevice, stream);

  // ================= MSA row attention =================
  run_ln(pair, row_lnzw, row_lnzb, W5, N * N, CZ, stream);
  bias_proj_k<<<(N * N) / 256, 256, 0, stream>>>(W5, row_wb, W9, N * N, CZ, 8);
  run_ln(msa, row_lnw, row_lnb, W0, S * N, CM, stream);
  { GemmP p = mk(W0, row_wq, nullptr, S * N, CM, CM, CM, CM); p.alpha = QS; p.Cb = W1; g_nn(p, 0, stream); }
  { GemmP p = mk(W0, row_wk, nullptr, S * N, CM, CM, CM, CM); p.Cb = W2; g_nn(p, 0, stream); }
  { GemmP p = mk(W0, row_wv, nullptr, S * N, CM, CM, CM, CM); p.Cb = W3; g_nn(p, 0, stream); }
  { GemmP p = mk(W0, row_wg, nullptr, S * N, CM, CM, CM, CM); p.bias = row_bg; p.Cf = W4; g_nn(p, 2, stream); }
  attn_k<1><<<dim3(N / 16, 8, S), 32, 0, stream>>>(W1, W2, W3, W4, W5, msa_mask, N, 1, W9, 8, N);
  { GemmP p = mk(W5, row_wo, nullptr, S * N, CM, CM, CM, CM); p.bias = row_bo; p.resid = msa; p.Cf = msa; g_nn(p, 0, stream); }

  // ================= MSA column attention =================
  perm_k<<<(unsigned)((MSA_E + 255) / 256), 256, 0, stream>>>(msa, W6, S, N, CM);
  run_ln(W6, col_lnw, col_lnb, W0, S * N, CM, stream);
  { GemmP p = mk(W0, col_wq, nullptr, S * N, CM, CM, CM, CM); p.alpha = QS; p.Cb = W1; g_nn(p, 0, stream); }
  { GemmP p = mk(W0, col_wk, nullptr, S * N, CM, CM, CM, CM); p.Cb = W2; g_nn(p, 0, stream); }
  { GemmP p = mk(W0, col_wv, nullptr, S * N, CM, CM, CM, CM); p.Cb = W3; g_nn(p, 0, stream); }
  { GemmP p = mk(W0, col_wg, nullptr, S * N, CM, CM, CM, CM); p.bias = col_bg; p.Cf = W4; g_nn(p, 2, stream); }
  attn_k<0><<<dim3(S / 16, 8, N), 32, 0, stream>>>(W1, W2, W3, W4, W5, msa_mask, 1, N, nullptr, 8, S);
  { GemmP p = mk(W5, col_wo, nullptr, S * N, CM, CM, CM, CM); p.bias = col_bo; p.resid = W6; p.Cf = W6; g_nn(p, 0, stream); }
  perm_k<<<(unsigned)((MSA_E + 255) / 256), 256, 0, stream>>>(W6, msa, N, S, CM);

  // ================= MSA transition =================
  run_ln(msa, mt_lnw, mt_lnb, W0, S * N, CM, stream);
  { GemmP p = mk(W0, mt_w1, nullptr, S * N, 4 * CM, CM, CM, 4 * CM); p.bias = mt_b1; p.Cb = W8; g_nn(p, 1, stream); }
  { GemmP p = mk(W8, mt_w2, nullptr, S * N, CM, 4 * CM, 4 * CM, CM); p.bias = mt_b2; p.rowscale = msa_mask; p.resid = msa; p.Cf = msa; g_nn(p, 0, stream); }

  // ================= Outer product mean =================
  run_ln(msa, op_lnw, op_lnb, W0, S * N, CM, stream);
  { GemmP p = mk(W0, op_w1, nullptr, S * N, 32, CM, CM, 32); p.bias = op_b1; p.rowscale = msa_mask; p.Cb = W1; g_nn(p, 0, stream); }
  { GemmP p = mk(W0, op_w2, nullptr, S * N, 32, CM, CM, 32); p.bias = op_b2; p.rowscale = msa_mask; p.Cb = W2; g_nn(p, 0, stream); }
  { GemmP p = mk(W1, nullptr, W2, 8192, 8192, S, 8192, 8192); p.Cb = W8; p.cbr = 8192; p.cbc = 1; g_atb(p, 1, stream); }
  opm_norm_k<<<(N * N) / 256, 256, 0, stream>>>(msa_mask, Wrs, S, N);
  { GemmP p = mk(W8, op_wo, nullptr, N * N, CZ, 1024, 0, CZ); p.bias = op_bo; p.rowscale = Wrs; p.resid = pair; p.Cf = pair; g_opm(p, stream); }

  // ================= Triangle multiplications =================
  auto tri_mul = [&](const float* lnw, const float* lnb,
                     const float* wap, const float* bap, const float* wag, const float* bag,
                     const float* wbp, const float* bbp, const float* wbg, const float* bbg,
                     const float* lnow, const float* lnob,
                     const float* wg, const float* bg, const float* wz, const float* bz,
                     bool outgoing) {
    run_ln(pair, lnw, lnb, W0, N * N, CZ, stream);
    { GemmP p = mk(W0, wap, nullptr, N * N, CZ, CZ, CZ, CZ); p.bias = bap; p.Cf = W6; g_nn(p, 0, stream); }
    { GemmP p = mk(W0, wag, nullptr, N * N, CZ, CZ, CZ, CZ); p.bias = bag; p.emul = W6; p.rowscale = pair_mask;
      p.Cb = W8; p.cbr = 1; p.cbc = N * N; g_nn(p, 2, stream); }                 // a: channel-major planes
    { GemmP p = mk(W0, wbp, nullptr, N * N, CZ, CZ, CZ, CZ); p.bias = bbp; p.Cf = W7; g_nn(p, 0, stream); }
    { GemmP p = mk(W0, wbg, nullptr, N * N, CZ, CZ, CZ, CZ); p.bias = bbg; p.emul = W7; p.rowscale = pair_mask;
      p.Cb = W8b; p.cbr = 1; p.cbc = N * N; g_nn(p, 2, stream); }                // b: channel-major planes
    { GemmP p = mk(W8, nullptr, W8b, N, N, N, N, N);
      p.Cf = W6; p.cfr = (long)N * CZ; p.cfc = CZ;
      p.sA = N * N; p.sB = N * N; p.sC = 1;
      if (outgoing) g_abt(p, CZ, stream);    // X_c = A_c @ B_c^T
      else          g_atb(p, CZ, stream); }  // X_c = A_c^T @ B_c
    run_ln(W6, lnow, lnob, W5, N * N, CZ, stream);
    { GemmP p = mk(W0, wg, nullptr, N * N, CZ, CZ, CZ, CZ); p.bias = bg; p.Cf = W4; g_nn(p, 2, stream); }
    { GemmP p = mk(W5, wz, nullptr, N * N, CZ, CZ, CZ, CZ); p.bias = bz; p.emul = W4; p.resid = pair; p.Cf = pair; g_nn(p, 0, stream); }
  };
  tri_mul(tmo_lnw, tmo_lnb, tmo_wap, tmo_bap, tmo_wag, tmo_bag, tmo_wbp, tmo_bbp, tmo_wbg, tmo_bbg,
          tmo_lnow, tmo_lnob, tmo_wg, tmo_bg, tmo_wz, tmo_bz, true);
  tri_mul(tmi_lnw, tmi_lnb, tmi_wap, tmi_bap, tmi_wag, tmi_bag, tmi_wbp, tmi_bbp, tmi_wbg, tmi_bbg,
          tmi_lnow, tmi_lnob, tmi_wg, tmi_bg, tmi_wz, tmi_bz, false);

  // ================= Triangle attentions =================
  auto tri_attn = [&](float* zbuf, long msb, long msk,
                      const float* lnw, const float* lnb, const float* wb,
                      const float* wq, const float* wk, const float* wv,
                      const float* wg, const float* bg, const float* wo, const float* bo) {
    run_ln(zbuf, lnw, lnb, W0, N * N, CZ, stream);
    bias_proj_k<<<(N * N) / 256, 256, 0, stream>>>(W0, wb, W9, N * N, CZ, 4);
    { GemmP p = mk(W0, wq, nullptr, N * N, CZ, CZ, CZ, CZ); p.alpha = QS; p.Cb = W1; g_nn(p, 0, stream); }
    { GemmP p = mk(W0, wk, nullptr, N * N, CZ, CZ, CZ, CZ); p.Cb = W2; g_nn(p, 0, stream); }
    { GemmP p = mk(W0, wv, nullptr, N * N, CZ, CZ, CZ, CZ); p.Cb = W3; g_nn(p, 0, stream); }
    { GemmP p = mk(W0, wg, nullptr, N * N, CZ, CZ, CZ, CZ); p.bias = bg; p.Cf = W4; g_nn(p, 2, stream); }
    attn_k<1><<<dim3(N / 16, 4, N), 32, 0, stream>>>(W1, W2, W3, W4, W5, pair_mask, msb, msk, W9, 4, N);
    { GemmP p = mk(W5, wo, nullptr, N * N, CZ, CZ, CZ, CZ); p.bias = bo; p.resid = zbuf; p.Cf = zbuf; g_nn(p, 0, stream); }
  };
  tri_attn(pair, N, 1, tas_lnw, tas_lnb, tas_wb, tas_wq, tas_wk, tas_wv, tas_wg, tas_bg, tas_wo, tas_bo);
  perm_k<<<(unsigned)((MSA_E + 255) / 256), 256, 0, stream>>>(pair, W6, N, N, CZ);
  tri_attn(W6, 1, N, tae_lnw, tae_lnb, tae_wb, tae_wq, tae_wk, tae_wv, tae_wg, tae_bg, tae_wo, tae_bo);
  perm_k<<<(unsigned)((MSA_E + 255) / 256), 256, 0, stream>>>(W6, pair, N, N, CZ);

  // ================= Pair transition =================
  run_ln(pair, pt_lnw, pt_lnb, W0, N * N, CZ, stream);
  { GemmP p = mk(W0, pt_w1, nullptr, N * N, 4 * CZ, CZ, CZ, 4 * CZ); p.bias = pt_b1; p.Cb = W8; g_nn(p, 1, stream); }
  { GemmP p = mk(W8, pt_w2, nullptr, N * N, CZ, 4 * CZ, 4 * CZ, CZ); p.bias = pt_b2; p.rowscale = pair_mask; p.resid = pair; p.Cf = pair; g_nn(p, 0, stream); }
}
